// AttentionNestedNERModel_5789615915510
// MI455X (gfx1250) — matrix operand order repaired and checked by reading, not compiled, not run on hardware
//
#include <hip/hip_runtime.h>
#include <hip/hip_bf16.h>

// Model dims
#define Sv 256
#define Bv 32
#define Ev 300
#define EP 320      // E padded to 32
#define Hv 256
#define H2v 512
#define Lv 4
#define Tv 9
#define DIN 1028
#define DINP 1056   // padded to 32

typedef __attribute__((ext_vector_type(16))) _Float16 v16h;
typedef __attribute__((ext_vector_type(8)))  float    v8f;
typedef __attribute__((ext_vector_type(4)))  unsigned u32x4;
typedef __attribute__((ext_vector_type(8)))  int      i32x8;
typedef __attribute__((ext_vector_type(4)))  int      i32x4;

#define DEV __device__ __forceinline__

DEV float sigf(float x) { return 1.f / (1.f + __expf(-x)); }

// A fragment (16x32 f16, M x K). lane = M (mod 16), dword v holds K pair
// (v/4)*16 + (lane/16)*8 + (v%4)*2. Rows are ldk halves apart (ldk even).
DEV v16h load_a16(const _Float16* base, int ldk, int m0, int k0, int lane) {
  int m  = m0 + (lane & 15);
  int kb = k0 + ((lane >> 4) << 3);
  const unsigned* p32 = (const unsigned*)(base + m * ldk + kb);
  const unsigned* q32 = (const unsigned*)(base + m * ldk + kb + 16);
  union { v16h h; unsigned u[8]; } f;
  f.u[0] = p32[0]; f.u[1] = p32[1]; f.u[2] = p32[2]; f.u[3] = p32[3];
  f.u[4] = q32[0]; f.u[5] = q32[1]; f.u[6] = q32[2]; f.u[7] = q32[3];
  return f.h;
}

// B fragment (32x16 f16, K x N) from pre-transposed weight WT[Kpad][N]:
// lane = N (mod 16), half s holds K = (s/8)*16 + (lane/16)*8 + s%8 (same as A).
DEV v16h load_b16(const _Float16* wt, int ldn, int k0, int n0, int lane) {
  int n  = n0 + (lane & 15);
  int kb = k0 + ((lane >> 4) << 3);
  v16h f;
#pragma unroll
  for (int s = 0; s < 8; ++s) {
    f[s]     = wt[(size_t)(kb + s) * ldn + n];
    f[8 + s] = wt[(size_t)(kb + 16 + s) * ldn + n];
  }
  return f;
}

DEV v8f wmma16(v16h a, v16h b, v8f c) {
  return __builtin_amdgcn_wmma_f32_16x16x32_f16(false, a, false, b, (short)0, c,
                                                false, false);
}

// Async global -> LDS 16B copy (ASYNCcnt). lds_off = byte offset in WG LDS.
DEV void async_g2l_b128(unsigned lds_off, const void* gaddr) {
  asm volatile("global_load_async_to_lds_b128 %0, %1, off"
               :: "v"(lds_off), "v"(gaddr) : "memory");
}
// Async LDS -> global 16B copy (ASYNCcnt).
DEV void async_l2g_b128(void* gaddr, unsigned lds_off) {
  asm volatile("global_store_async_from_lds_b128 %0, %1, off"
               :: "v"(gaddr), "v"(lds_off) : "memory");
}
DEV void wait_asynccnt0() {
  asm volatile("s_wait_asynccnt 0" ::: "memory");
}

// ---------------- prep kernels ----------------

// dst[Kpad][Npad] f16 = transpose(src[N][K]) with zero pad. If perm, dest col j
// maps to source row (j&3)*512 + (j>>2)  (gate-interleave for decoder weights).
__global__ void k_cvt_T(const float* __restrict__ src, _Float16* __restrict__ dst,
                        int N, int K, int Kpad, int Npad, int perm) {
  int idx = blockIdx.x * blockDim.x + threadIdx.x;
  int total = Kpad * Npad;
  if (idx >= total) return;
  int k = idx / Npad, j = idx % Npad;
  int n = perm ? ((j & 3) * 512 + (j >> 2)) : j;
  float v = (k < K && n < N) ? src[(size_t)n * K + k] : 0.f;
  dst[idx] = (_Float16)v;
}

__global__ void k_perm_bias(const float* __restrict__ src, float* __restrict__ dst) {
  int j = blockIdx.x * blockDim.x + threadIdx.x;
  if (j < 4 * H2v) dst[j] = src[(j & 3) * 512 + (j >> 2)];
}

// x[s][b][EP] f16 = emb[seqs[b][s]] (zero padded E->EP)
__global__ void k_embed(const int* __restrict__ seqs, const float* __restrict__ emb,
                        _Float16* __restrict__ x) {
  int sb = blockIdx.x;            // s*B + b
  int s = sb >> 5, b = sb & 31;
  int tok = seqs[b * Sv + s];
  const float* e = emb + (size_t)tok * Ev;
  for (int k = threadIdx.x; k < EP; k += blockDim.x)
    x[(size_t)sb * EP + k] = (_Float16)(k < Ev ? e[k] : 0.f);
}

// ---------------- encoder (one WG per direction) ----------------

#define ENC_G_OFF  0
#define ENC_C_OFF  131072
#define ENC_H_OFF  (131072 + 32768)
#define ENC_X_OFF  (131072 + 32768 + 16384)

__global__ void __launch_bounds__(256) k_encoder(
    const _Float16* __restrict__ x,
    const _Float16* __restrict__ WihT_f, const _Float16* __restrict__ WhhT_f,
    const float* __restrict__ bias_f,
    const _Float16* __restrict__ WihT_b, const _Float16* __restrict__ WhhT_b,
    const float* __restrict__ bias_b,
    _Float16* __restrict__ h_enc) {
  extern __shared__ char smem[];
  float*    G  = (float*)(smem + ENC_G_OFF);   // [32][1024] 128K
  float*    C  = (float*)(smem + ENC_C_OFF);   // [32][256]  32K
  _Float16* Hs = (_Float16*)(smem + ENC_H_OFF); // [32][256] 16K
  _Float16* Xs = (_Float16*)(smem + ENC_X_OFF); // [32][320] 20K

  const int dir = blockIdx.x;
  const _Float16* WihT = dir ? WihT_b : WihT_f;
  const _Float16* WhhT = dir ? WhhT_b : WhhT_f;
  const float*    bias = dir ? bias_b : bias_f;
  int tid = threadIdx.x, lane = tid & 31, wave = tid >> 5;

  for (int i = tid; i < Bv * Hv; i += 256) { C[i] = 0.f; Hs[i] = (_Float16)0.f; }
  __syncthreads();

  for (int t = 0; t < Sv; ++t) {
    int s = dir ? (Sv - 1 - t) : t;
    const _Float16* xrow = x + (size_t)s * Bv * EP;

#if __has_builtin(__builtin_amdgcn_tensor_load_to_lds) && __has_builtin(__builtin_amdgcn_s_wait_tensorcnt)
    // Tensor Data Mover: DMA the whole 20KB X_t tile into LDS in one op.
    if (tid == 0) {
      unsigned long long ga = (unsigned long long)(const void*)xrow;
      u32x4 g0;
      g0[0] = 1u;                                   // count=1, user descriptor
      g0[1] = (unsigned)ENC_X_OFF;                  // lds_addr (bytes)
      g0[2] = (unsigned)(ga & 0xffffffffu);         // global_addr[31:0]
      g0[3] = (unsigned)((ga >> 32) & 0x1ffffffu)   // global_addr[56:32]
              | (2u << 30);                         // type = 2 ("image")
      i32x8 g1;
      const unsigned td0 = Bv * EP;                 // 10240 elements, 1-D tile
      g1[0] = (int)(1u << 16);                      // data_size = 2 bytes
      g1[1] = (int)((td0 & 0xffffu) << 16);         // tensor_dim0[15:0]
      g1[2] = (int)((td0 >> 16) & 0xffffu);         // tensor_dim0[31:16]
      g1[3] = (int)((td0 & 0xffffu) << 16);         // tile_dim0
      g1[4] = 0;                                    // tile_dim1/2 unused
      g1[5] = (int)td0;                             // tensor_dim0_stride
      g1[6] = 0;
      g1[7] = 0;
      i32x4 z4 = {0, 0, 0, 0};
      i32x8 z8 = {0, 0, 0, 0, 0, 0, 0, 0};
      __builtin_amdgcn_tensor_load_to_lds(g0, g1, z4, z4, z8, 0);
    }
    __builtin_amdgcn_s_wait_tensorcnt(0);
#else
    for (int i = tid; i < Bv * EP / 8; i += 256)
      ((uint4*)Xs)[i] = ((const uint4*)xrow)[i];
#endif
    __syncthreads();

    // G = Xs*Wih^T + Hs*Whh^T + b ; each wave owns 8 N-tiles x 2 M-tiles
    for (int nt = wave * 8; nt < wave * 8 + 8; ++nt) {
      int n0 = nt * 16;
      float bv = bias[n0 + (lane & 15)];
      v8f acc0 = {bv, bv, bv, bv, bv, bv, bv, bv};
      v8f acc1 = acc0;
      for (int kt = 0; kt < EP / 32; ++kt) {
        __builtin_prefetch(WihT + (size_t)(kt * 32 + 32 + lane) * 1024 + n0, 0, 1);
        v16h bf = load_b16(WihT, 1024, kt * 32, n0, lane);
        v16h a0 = load_a16(Xs, EP, 0, kt * 32, lane);
        v16h a1 = load_a16(Xs, EP, 16, kt * 32, lane);
        acc0 = wmma16(a0, bf, acc0);
        acc1 = wmma16(a1, bf, acc1);
      }
      for (int kt = 0; kt < Hv / 32; ++kt) {
        v16h bf = load_b16(WhhT, 1024, kt * 32, n0, lane);
        v16h a0 = load_a16(Hs, Hv, 0, kt * 32, lane);
        v16h a1 = load_a16(Hs, Hv, 16, kt * 32, lane);
        acc0 = wmma16(a0, bf, acc0);
        acc1 = wmma16(a1, bf, acc1);
      }
      int n = n0 + (lane & 15), mh = (lane >> 4) * 8;
#pragma unroll
      for (int v = 0; v < 8; ++v) {
        G[(mh + v) * 1024 + n]      = acc0[v];
        G[(16 + mh + v) * 1024 + n] = acc1[v];
      }
    }
    __syncthreads();

    // LSTM cell (gate order i,f,g,o at n, 256+n, 512+n, 768+n)
    for (int u = tid; u < Bv * Hv; u += 256) {
      int b = u >> 8, n = u & 255;
      float gi = G[b * 1024 + n],       gf = G[b * 1024 + 256 + n];
      float gg = G[b * 1024 + 512 + n], go = G[b * 1024 + 768 + n];
      float c = sigf(gf) * C[u] + sigf(gi) * tanhf(gg);
      C[u] = c;
      float h = sigf(go) * tanhf(c);
      Hs[u] = (_Float16)h;
      h_enc[((size_t)s * Bv + b) * H2v + dir * Hv + n] = (_Float16)h;
    }
    __syncthreads();
  }
}

// ---------------- Wh = h_enc * W_att^T  (parallel GEMM) ----------------

__global__ void __launch_bounds__(256) k_att_proj(
    const _Float16* __restrict__ h_enc, const _Float16* __restrict__ WattT,
    _Float16* __restrict__ Wh) {
  int tid = threadIdx.x, lane = tid & 31, wave = tid >> 5;
  int m0 = blockIdx.x * 32;
  for (int i = 0; i < 4; ++i) {
    int n0 = (wave * 4 + i) * 16;
    v8f acc0 = {}, acc1 = {};
    for (int kt = 0; kt < H2v / 32; ++kt) {
      v16h bf = load_b16(WattT, H2v, kt * 32, n0, lane);
      v16h a0 = load_a16(h_enc, H2v, m0, kt * 32, lane);
      v16h a1 = load_a16(h_enc, H2v, m0 + 16, kt * 32, lane);
      acc0 = wmma16(a0, bf, acc0);
      acc1 = wmma16(a1, bf, acc1);
    }
    int n = n0 + (lane & 15), mh = (lane >> 4) * 8;
#pragma unroll
    for (int v = 0; v < 8; ++v) {
      Wh[(size_t)(m0 + mh + v) * H2v + n]      = (_Float16)acc0[v];
      Wh[(size_t)(m0 + 16 + mh + v) * H2v + n] = (_Float16)acc1[v];
    }
  }
}

// ---------------- decoder (single persistent WG, 1024 serial steps) --------

#define DEC_XIN_OFF 0
#define DEC_S0_OFF  67584
#define DEC_S1_OFF  (67584 + 32768)
#define DEC_C_OFF   (67584 + 65536)
#define DEC_G_OFF   (67584 + 131072)
#define DEC_ATT_OFF (67584 + 196608)

__global__ void __launch_bounds__(256) k_decoder(
    const _Float16* __restrict__ h_enc, const _Float16* __restrict__ Wh,
    const _Float16* __restrict__ WihT, const _Float16* __restrict__ WhhT,
    const float* __restrict__ bperm, _Float16* __restrict__ outs) {
  extern __shared__ char smem[];
  _Float16* xin = (_Float16*)(smem + DEC_XIN_OFF); // [32][1056]
  _Float16* S0  = (_Float16*)(smem + DEC_S0_OFF);  // [32][512]
  _Float16* S1  = (_Float16*)(smem + DEC_S1_OFF);  // [32][512]
  float*    C   = (float*)(smem + DEC_C_OFF);      // [32][512] f32
  float*    Gp  = (float*)(smem + DEC_G_OFF);      // [32][512] f32
  float*    att = (float*)(smem + DEC_ATT_OFF);    // [32][256] f32
  int tid = threadIdx.x, lane = tid & 31, wave = tid >> 5;

  for (int i = tid; i < Bv * H2v; i += 256) {
    C[i] = 0.f; S0[i] = (_Float16)0.f; S1[i] = (_Float16)0.f;
  }
  for (int i = tid; i < Bv * DINP; i += 256) xin[i] = (_Float16)0.f;
  __syncthreads();

  for (int l = 0; l < Lv; ++l) {
    for (int i = tid; i < Bv * 4; i += 256) {
      int b = i >> 2, k = i & 3;
      xin[b * DINP + 1024 + k] = (_Float16)(k == l ? 1.f : 0.f);
    }
    __syncthreads();

    for (int t = 0; t < Sv; ++t) {
      int st = l * Sv + t;
      _Float16* Sp = (st & 1) ? S1 : S0;   // previous hidden state
      _Float16* Sn = (st & 1) ? S0 : S1;   // new hidden state

      // stage h_t into xin cols [512,1024) via async global->LDS DMA
      const _Float16* hrow = h_enc + (size_t)t * Bv * H2v;
      for (int ch = tid; ch < Bv * H2v / 8; ch += 256) {
        int b = ch >> 6, k8 = (ch & 63) * 8;
        unsigned lds = (unsigned)(DEC_XIN_OFF + (b * DINP + 512 + k8) * 2);
        async_g2l_b128(lds, hrow + ch * 8);
      }
      wait_asynccnt0();

      // score[b][j] = Wh[j,b,:] . Sp[b,:]   (thread = j)
      {
        int j = tid;
        const _Float16* wrow = Wh + (size_t)j * Bv * H2v;
        for (int b = 0; b < Bv; ++b) {
          const _Float16* wr = wrow + b * H2v;
          const _Float16* sr = Sp + b * H2v;
          float acc = 0.f;
          for (int o = 0; o < H2v; ++o) acc += (float)wr[o] * (float)sr[o];
          att[b * Sv + j] = acc;
        }
      }
      __syncthreads();

      // softmax rows (one lane per batch row)
      if (tid < Bv) {
        float* row = att + tid * Sv;
        float mx = -1e30f;
        for (int j = 0; j < Sv; ++j) mx = fmaxf(mx, row[j]);
        float sum = 0.f;
        for (int j = 0; j < Sv; ++j) { float e = __expf(row[j] - mx); row[j] = e; sum += e; }
        float inv = 1.f / sum;
        for (int j = 0; j < Sv; ++j) row[j] *= inv;
      }
      __syncthreads();

      // ctx into xin cols [0,512)
      {
        int b = tid >> 3, h0 = (tid & 7) * 64;
        for (int h = h0; h < h0 + 64; ++h) {
          float acc = 0.f;
          for (int j = 0; j < Sv; ++j)
            acc += att[b * Sv + j] * (float)h_enc[((size_t)j * Bv + b) * H2v + h];
          xin[b * DINP + h] = (_Float16)acc;
        }
      }
      __syncthreads();

      // gates in 4 N-passes of 512 interleaved columns (= 128 hidden units)
      for (int p = 0; p < 4; ++p) {
        for (int i = 0; i < 4; ++i) {
          int ntl = wave * 4 + i;
          int n0g = p * 512 + ntl * 16;
          float bv = bperm[n0g + (lane & 15)];
          v8f acc0 = {bv, bv, bv, bv, bv, bv, bv, bv};
          v8f acc1 = acc0;
          for (int kt = 0; kt < DINP / 32; ++kt) {
            __builtin_prefetch(WihT + (size_t)(kt * 32 + 32 + lane) * 2048 + n0g, 0, 1);
            v16h bf = load_b16(WihT, 2048, kt * 32, n0g, lane);
            v16h a0 = load_a16(xin, DINP, 0, kt * 32, lane);
            v16h a1 = load_a16(xin, DINP, 16, kt * 32, lane);
            acc0 = wmma16(a0, bf, acc0);
            acc1 = wmma16(a1, bf, acc1);
          }
          for (int kt = 0; kt < H2v / 32; ++kt) {
            v16h bf = load_b16(WhhT, 2048, kt * 32, n0g, lane);
            v16h a0 = load_a16(Sp, H2v, 0, kt * 32, lane);
            v16h a1 = load_a16(Sp, H2v, 16, kt * 32, lane);
            acc0 = wmma16(a0, bf, acc0);
            acc1 = wmma16(a1, bf, acc1);
          }
          int n = ntl * 16 + (lane & 15), mh = (lane >> 4) * 8;
#pragma unroll
          for (int v = 0; v < 8; ++v) {
            Gp[(mh + v) * 512 + n]      = acc0[v];
            Gp[(16 + mh + v) * 512 + n] = acc1[v];
          }
        }
        __syncthreads();
        // local col 4*nn+{0,1,2,3} = gates i,f,g,o of hidden p*128+nn
        for (int u = tid; u < Bv * 128; u += 256) {
          int b = u >> 7, nn = u & 127;
          int nh = p * 128 + nn;
          float gi = Gp[b * 512 + 4 * nn + 0], gf = Gp[b * 512 + 4 * nn + 1];
          float gg = Gp[b * 512 + 4 * nn + 2], go = Gp[b * 512 + 4 * nn + 3];
          float c = sigf(gf) * C[b * H2v + nh] + sigf(gi) * tanhf(gg);
          C[b * H2v + nh] = c;
          Sn[b * H2v + nh] = (_Float16)(sigf(go) * tanhf(c));
        }
        __syncthreads();
      }

      // emit decoder hidden state straight from LDS via async LDS->global DMA
      {
        unsigned sn_off = (unsigned)((st & 1) ? DEC_S0_OFF : DEC_S1_OFF);
        _Float16* orow = outs + (size_t)st * Bv * H2v;
        for (int ch = tid; ch < Bv * H2v / 8; ch += 256)
          async_l2g_b128(orow + ch * 8, sn_off + (unsigned)(ch * 16));
        wait_asynccnt0();
      }
      __syncthreads();
    }
  }
}

// ---------------- tag projection ----------------

__global__ void __launch_bounds__(256) k_tag(
    const _Float16* __restrict__ outs, const _Float16* __restrict__ WtagT,
    const float* __restrict__ b_tag, float* __restrict__ out) {
  int tid = threadIdx.x, lane = tid & 31, wave = tid >> 5;
  int m0 = blockIdx.x * 128 + wave * 16;
  v8f acc = {};
  for (int kt = 0; kt < H2v / 32; ++kt) {
    v16h bf = load_b16(WtagT, 16, kt * 32, 0, lane);
    v16h a  = load_a16(outs, H2v, m0, kt * 32, lane);
    acc = wmma16(a, bf, acc);
  }
  int n = lane & 15, mh = (lane >> 4) * 8;
  if (n < Tv) {
    float bt = b_tag[n];
#pragma unroll
    for (int v = 0; v < 8; ++v)
      out[(size_t)(m0 + mh + v) * Tv + n] = acc[v] + bt;
  }
}

// ---------------- launcher ----------------

extern "C" void kernel_launch(void* const* d_in, const int* in_sizes, int n_in,
                              void* d_out, int out_size, void* d_ws, size_t ws_size,
                              hipStream_t stream) {
  const int*   seqs  = (const int*)d_in[0];
  const float* emb   = (const float*)d_in[1];
  const float* Wih_f = (const float*)d_in[2];
  const float* Whh_f = (const float*)d_in[3];
  const float* b_f   = (const float*)d_in[4];
  const float* Wih_b = (const float*)d_in[5];
  const float* Whh_b = (const float*)d_in[6];
  const float* b_b   = (const float*)d_in[7];
  const float* W_att = (const float*)d_in[8];
  const float* dWih  = (const float*)d_in[9];
  const float* dWhh  = (const float*)d_in[10];
  const float* db    = (const float*)d_in[11];
  const float* W_tag = (const float*)d_in[12];
  const float* btag  = (const float*)d_in[13];

  char* ws = (char*)d_ws;
  _Float16* x       = (_Float16*)(ws + 0);          // S*B*320        5,242,880B
  _Float16* WihT_f  = (_Float16*)(ws + 5242880);    // 320*1024         655,360
  _Float16* WhhT_f  = (_Float16*)(ws + 5898240);    // 256*1024         524,288
  _Float16* WihT_b  = (_Float16*)(ws + 6422528);
  _Float16* WhhT_b  = (_Float16*)(ws + 7077888);
  _Float16* h_enc   = (_Float16*)(ws + 7602176);    // S*B*512        8,388,608
  _Float16* WattT   = (_Float16*)(ws + 15990784);   // 512*512          524,288
  _Float16* Wh      = (_Float16*)(ws + 16515072);   // S*B*512        8,388,608
  _Float16* dWihT   = (_Float16*)(ws + 24903680);   // 1056*2048      4,325,376
  _Float16* dWhhT   = (_Float16*)(ws + 29229056);   // 512*2048       2,097,152
  float*    bperm   = (float*)(ws + 31326208);      // 2048 f32           8,192
  _Float16* WtagT   = (_Float16*)(ws + 31334400);   // 512*16            16,384
  _Float16* outs    = (_Float16*)(ws + 31350784);   // L*S*B*512     33,554,432

  auto cvt = [&](const float* s, _Float16* d, int N, int K, int Kp, int Np, int perm) {
    int total = Kp * Np;
    k_cvt_T<<<(total + 255) / 256, 256, 0, stream>>>(s, d, N, K, Kp, Np, perm);
  };
  cvt(Wih_f, WihT_f, 1024, Ev, EP, 1024, 0);
  cvt(Whh_f, WhhT_f, 1024, Hv, Hv, 1024, 0);
  cvt(Wih_b, WihT_b, 1024, Ev, EP, 1024, 0);
  cvt(Whh_b, WhhT_b, 1024, Hv, Hv, 1024, 0);
  cvt(W_att, WattT, H2v, H2v, H2v, H2v, 0);
  cvt(dWih, dWihT, 2048, DIN, DINP, 2048, 1);
  cvt(dWhh, dWhhT, 2048, H2v, H2v, 2048, 1);
  cvt(W_tag, WtagT, Tv, H2v, H2v, 16, 0);
  k_perm_bias<<<8, 256, 0, stream>>>(db, bperm);

  k_embed<<<Sv * Bv, 64, 0, stream>>>(seqs, emb, x);

  // encoder: one WG per direction, 200,704B dynamic LDS
  k_encoder<<<2, 256, 200704, stream>>>(x, WihT_f, WhhT_f, b_f,
                                        WihT_b, WhhT_b, b_b, h_enc);

  k_att_proj<<<Sv * Bv / 32, 256, 0, stream>>>(h_enc, WattT, Wh);

  // decoder: persistent single WG, 296,960B dynamic LDS
  k_decoder<<<1, 256, 296960, stream>>>(h_enc, Wh, dWihT, dWhhT, bperm, outs);

  k_tag<<<Lv * Sv * Bv / 128, 256, 0, stream>>>(outs, WtagT, btag, (float*)d_out);
}
